// Node2VecModel_44985487458536
// MI455X (gfx1250) — matrix operand order repaired and checked by reading, not compiled, not run on hardware
//
#include <hip/hip_runtime.h>
#include <stdint.h>

// ---------------------------------------------------------------------------
// Node2Vec skip-gram NEG loss, MI455X (gfx1250).
//
// Per pair b: gather rows emb[v], emb[u], emb[neg[b][0..19]] (22 x 512B) via
// TDM gather-mode descriptors (3 x tensor_load_to_lds, 8+8+6 rows) into LDS,
// then compute the 21 dot products with v_wmma_f32_16x16x32_f16 (B = v-row
// broadcast across columns), then softplus + reduce.
// Memory-bound: ~738 MB of gathers -> ~32 us at 23.3 TB/s; the code path is
// tuned so per-wave VALU work stays well under that bound.
//
// Tile-1 rows >= 21 read a per-wave zero pad row (clamped index) so fragment
// building is branch-free: ds_load_b128 + v_cvt_pk_f16_f32 only.
// ---------------------------------------------------------------------------

typedef __attribute__((ext_vector_type(16))) _Float16 v16h;
typedef __attribute__((ext_vector_type(8)))  float    v8f;
typedef __attribute__((ext_vector_type(4)))  unsigned u32x4;
typedef __attribute__((ext_vector_type(8)))  int      i32x8;
typedef __attribute__((ext_vector_type(4)))  int      i32x4;

#define EMB    128
#define KNEG   20
#define WAVES  4          // 4 waves/block, one (u,v) pair per wave
#define ROWS   23         // v + (u + 20 neg) + 1 zero pad row

// softplus(y) = log(1+exp(y)), stable; note -log_sigmoid(x) == softplus(-x).
// Uses hw transcendentals (v_exp_f32 / v_log_f32), ~6 VALU ops.
__device__ __forceinline__ float softplus(float y) {
  return fmaxf(y, 0.0f) + __logf(1.0f + __expf(-fabsf(y)));
}

// TDM gather-mode load: pull `nrows` rows of 128 fp32 (indices in idx_lo/idx_hi,
// 32-bit index mode) from `base` into LDS at byte offset `lds_off`, packed
// consecutively. D# packing per CDNA5 ISA.
// 6-arg builtin: (u32x4 g0, i32x8 g1, i32x4 g2, i32x4 g3, i32x8, i32 cpol)
__device__ __forceinline__ void tdm_gather_rows(const float* base,
                                                unsigned lds_off,
                                                i32x4 idx_lo, i32x4 idx_hi,
                                                int nrows, unsigned vocab) {
  unsigned long long ga = (unsigned long long)(uintptr_t)base;
  u32x4 g0;
  g0[0] = 1u            // count = 1 (valid descriptor)
        | (1u << 30)    // gather_index_size = 32-bit
        | (1u << 31);   // gather_mode = 1
  g0[1] = lds_off;                                        // lds_addr
  g0[2] = (unsigned)ga;                                   // global_addr[31:0]
  g0[3] = (unsigned)((ga >> 32) & 0x01FFFFFFu)            // global_addr[56:32]
        | (2u << 30);                                     // type = 2 (image)
  i32x8 g1;
  g1[0] = (int)(2u << 16);                                // data_size = 4B
  g1[1] = (int)((unsigned)EMB << 16);                     // tensor_dim0 lo16 = 128
  g1[2] = (int)((vocab & 0xFFFFu) << 16);                 // dim0 hi16=0 | tensor_dim1 lo16
  g1[3] = (int)((vocab >> 16) | ((unsigned)EMB << 16));   // tensor_dim1 hi16 | tile_dim0=128
  g1[4] = nrows;                                          // tile_dim1 = #valid indices
  g1[5] = EMB;                                            // tensor_dim0_stride lo32 = 128
  g1[6] = 0;                                              // stride hi16 | dim1_stride lo (ignored)
  g1[7] = 0;
  i32x8 gx = {};                                          // unused extra operand
  __builtin_amdgcn_tensor_load_to_lds(g0, g1, idx_lo, idx_hi, gx, 0);
}

__global__ __launch_bounds__(32 * WAVES)
void n2v_neg_loss_kernel(const int* __restrict__ u_idx,
                         const int* __restrict__ v_idx,
                         const int* __restrict__ neg_idx,
                         const float* __restrict__ emb,
                         float* __restrict__ out,
                         int npairs, unsigned vocab) {
  __shared__ __align__(1024) float smem[WAVES][ROWS * EMB];

  const int lane  = (int)(threadIdx.x & 31u);
  const int wave  = (int)(threadIdx.x >> 5u);
  const int b     = (int)blockIdx.x * WAVES + wave;
  if (b >= npairs) return;   // wave-uniform; grid sized exactly in practice

  float* sm = smem[wave];
  // Generic shared pointer: addr[31:0] IS the LDS byte offset (aperture map).
  const unsigned lds_base = (unsigned)(uintptr_t)sm;

  // Wave-uniform index loads (scalar path expected).
  const int  iv = v_idx[b];
  const int  iu = u_idx[b];
  const int* nb = neg_idx + (long)b * KNEG;

  // LDS row layout: row 0 = v; rows 1..21 = [u, neg0..neg19] (A rows 0..20);
  // row 22 = zero pad (A row 21), read by all out-of-range tile-1 rows.
  i32x4 q0, q1;
  q0[0] = iv;     q0[1] = iu;     q0[2] = nb[0];  q0[3] = nb[1];
  q1[0] = nb[2];  q1[1] = nb[3];  q1[2] = nb[4];  q1[3] = nb[5];
  tdm_gather_rows(emb, lds_base,                 q0, q1, 8, vocab);
  q0[0] = nb[6];  q0[1] = nb[7];  q0[2] = nb[8];  q0[3] = nb[9];
  q1[0] = nb[10]; q1[1] = nb[11]; q1[2] = nb[12]; q1[3] = nb[13];
  tdm_gather_rows(emb, lds_base +  8u * EMB * 4, q0, q1, 8, vocab);
  q0[0] = nb[14]; q0[1] = nb[15]; q0[2] = nb[16]; q0[3] = nb[17];
  q1[0] = nb[18]; q1[1] = nb[19]; q1[2] = 0;      q1[3] = 0;
  tdm_gather_rows(emb, lds_base + 16u * EMB * 4, q0, q1, 6, vocab);

  // Zero the pad row (A row 21) while the TDM gathers are in flight.
  {
    float4 z4 = make_float4(0.f, 0.f, 0.f, 0.f);
    reinterpret_cast<float4*>(sm + 22 * EMB)[lane] = z4;   // 32 x 16B = 512B
  }

  __builtin_amdgcn_s_wait_tensorcnt(0);

  const float* smV = sm;          // v row (B matrix, broadcast over columns)
  const float* smA = sm + EMB;    // A rows: 0=u, 1..20=negatives, 21=zeros

  const int lhalf = lane >> 4;    // 0: lanes 0-15, 1: lanes 16-31
  const int row0  = (lane & 15);              // tile-0 row for this lane
  const int row1c = min(16 + (lane & 15), 21);// tile-1 row, clamped to pad row

  v8f acc0 = {};                  // scores for A rows 0..15
  v8f acc1 = {};                  // scores for A rows 16..20 (rest hit pad=0)

#pragma unroll
  for (int c = 0; c < 4; ++c) {   // K chunks of 32 over D=128
    const int kb = c * 32;

    // B 32x16 f16 fragment: lanes 0-15 hold K {kb+0..7, kb+16..23}, lanes 16-31 K {kb+8..15, kb+24..31},
    // identical across the 16 columns (broadcast of the v row).
    v16h bf;
    const float* vb = smV + kb + lhalf * 8;
#pragma unroll
    for (int j = 0; j < 8; ++j) { bf[j] = (_Float16)vb[j]; bf[8 + j] = (_Float16)vb[16 + j]; }

    // A 16x32 f16 fragments: lane L<16 holds row L, K {kb+0..7, kb+16..23};
    // lane L>=16 holds row L-16, K {kb+8..15, kb+24..31}. Branch-free.
    {
      const float* ar = smA + row0 * EMB + kb + lhalf * 8;
      v16h af;
#pragma unroll
      for (int j = 0; j < 8; ++j) {
        af[j]     = (_Float16)ar[j];
        af[8 + j] = (_Float16)ar[16 + j];
      }
      acc0 = __builtin_amdgcn_wmma_f32_16x16x32_f16(
          false, af, false, bf, (short)0, acc0, false, false);
    }
    {
      const float* ar = smA + row1c * EMB + kb + lhalf * 8;
      v16h af;
#pragma unroll
      for (int j = 0; j < 8; ++j) {
        af[j]     = (_Float16)ar[j];
        af[8 + j] = (_Float16)ar[16 + j];
      }
      acc1 = __builtin_amdgcn_wmma_f32_16x16x32_f16(
          false, af, false, bf, (short)0, acc1, false, false);
    }
  }

  // D 16x16 f32 layout: VGPR r, lanes 0-15 -> M=r, lanes 16-31 -> M=r+8.
  // All columns identical within a half-wave.
  // loss = softplus(-s_pos) + sum_k softplus(s_neg_k); branch-free epilogue.
  float part = 0.0f;
#pragma unroll
  for (int r = 0; r < 8; ++r) {
    const int   m0 = lhalf * 8 + r;             // A row index in tile 0
    const float y0 = (m0 == 0) ? -acc0[r] : acc0[r];  // one cndmask
    part += softplus(y0);
    const int   m1   = 16 + m0;                 // A row index in tile 1
    const float mask = (m1 < 21) ? 1.0f : 0.0f; // pad rows give s==0, masked out
    part += mask * softplus(acc1[r]);
  }
  // halves hold duplicated values -> one xor-16 combine gives the full sum
  part += __shfl_xor(part, 16);
  if (lane == 0) out[b] = part;
}

extern "C" void kernel_launch(void* const* d_in, const int* in_sizes, int n_in,
                              void* d_out, int out_size, void* d_ws, size_t ws_size,
                              hipStream_t stream) {
  const int*   u   = (const int*)d_in[0];
  const int*   v   = (const int*)d_in[1];
  const int*   neg = (const int*)d_in[2];
  const float* emb = (const float*)d_in[3];
  float*       out = (float*)d_out;

  const int      B = in_sizes[0];
  const unsigned V = (unsigned)(in_sizes[3] / EMB);

  const int blocks = (B + WAVES - 1) / WAVES;
  n2v_neg_loss_kernel<<<blocks, 32 * WAVES, 0, stream>>>(u, v, neg, emb, out, B, V);
}
